// MHSA_7791070675723
// MI455X (gfx1250) — matrix operand order repaired and checked by reading, not compiled, not run on hardware
//
#include <hip/hip_runtime.h>

#define BB 4
#define SS 2048
#define DD 1024
#define HH 16
#define HDIM 64
#define MM (BB * SS)   // 8192

typedef __bf16 bf16;
typedef __attribute__((ext_vector_type(16))) __bf16 bf16x16;
typedef __attribute__((ext_vector_type(8)))  __bf16 bf16x8;
typedef __attribute__((ext_vector_type(8)))  float  f32x8;
typedef __attribute__((ext_vector_type(4)))  int    v4i;

union Frag16 { bf16x16 v; bf16x8 h[2]; };

#if defined(__HIP_DEVICE_COMPILE__) && __has_builtin(__builtin_amdgcn_global_load_async_to_lds_b128)
#define USE_ASYNC 1
#else
#define USE_ASYNC 0
#endif

static __device__ __forceinline__ f32x8 wmma_bf16(bf16x16 a, bf16x16 b, f32x8 c) {
  // D = A(16x32 bf16) x B(32x16 bf16) + C(16x16 f32)
  return __builtin_amdgcn_wmma_f32_16x16x32_bf16(false, a, false, b, (short)0, c,
                                                 false, false);
}

static __device__ __forceinline__ bf16x8 ld8(const bf16* p) {
  return *(const bf16x8*)p;
}

static __device__ __forceinline__ f32x8 fzero8() {
  f32x8 z = {0.f, 0.f, 0.f, 0.f, 0.f, 0.f, 0.f, 0.f};
  return z;
}

#if USE_ASYNC
typedef __attribute__((address_space(1))) v4i* gbl_v4i_p;
typedef __attribute__((address_space(3))) v4i* lds_v4i_p;
static __device__ __forceinline__ void async_cp_b128(const bf16* g, bf16* l) {
  __builtin_amdgcn_global_load_async_to_lds_b128((gbl_v4i_p)g, (lds_v4i_p)l, 0, 0);
}
#endif

// ---------------------------------------------------------------- converts
__global__ void cvt_x_kernel(const float* __restrict__ x, bf16* __restrict__ xb, int n) {
  int i = blockIdx.x * blockDim.x + threadIdx.x;
  if (i < n) xb[i] = (bf16)x[i];
}

// Wq/Wk/Wv [H, D, HD] fp32  ->  Wt [3*D, D] bf16 (row = which*1024 + h*64 + j, col = d)
__global__ void cvt_wqkv_kernel(const float* __restrict__ Wq, const float* __restrict__ Wk,
                                const float* __restrict__ Wv, bf16* __restrict__ Bt) {
  const int per = HH * DD * HDIM;           // 1,048,576
  int i = blockIdx.x * blockDim.x + threadIdx.x;
  if (i >= 3 * per) return;
  int which = i / per;
  int rem   = i - which * per;
  int h  = rem / (DD * HDIM);
  int r2 = rem - h * (DD * HDIM);
  int d  = r2 / HDIM;
  int j  = r2 - d * HDIM;
  const float* src = (which == 0) ? Wq : ((which == 1) ? Wk : Wv);
  Bt[(size_t)(which * DD + h * HDIM + j) * DD + d] = (bf16)src[rem];
}

// Wo [D, D] -> Wot [D, D] transposed bf16
__global__ void cvt_wo_kernel(const float* __restrict__ Wo, bf16* __restrict__ Wot) {
  int i = blockIdx.x * blockDim.x + threadIdx.x;
  if (i >= DD * DD) return;
  int d = i / DD, e = i - d * DD;
  Wot[(size_t)e * DD + d] = (bf16)Wo[i];
}

// ---------------------------------------------------------------- QKV GEMM
// 64x64 tile per wave: 16 WMMAs per 32-deep k-step vs 8 fragment loads.
//   cols [0,1024)   -> Q  [B,H,S,HD] bf16, scaled by 1/64 (scores/HD folded in)
//   cols [1024,2048)-> K  [B,H,S,HD] bf16
//   cols [2048,3072)-> Vt [B,H,HD,S] bf16 (transposed for PV B-operand)
__global__ __launch_bounds__(256) void gemm_qkv_kernel(
    const bf16* __restrict__ A, const bf16* __restrict__ Bt,
    bf16* __restrict__ Qb, bf16* __restrict__ Kb, bf16* __restrict__ Vt) {
  const int lane = threadIdx.x & 31;
  const int wave = threadIdx.x >> 5;
  const int half = lane >> 4;
  const int l16  = lane & 15;
  const int m0 = blockIdx.x * 64;
  const int n0 = (blockIdx.y * 8 + wave) * 64;
  const int aklo = half * 16;   // A-operand per-lane K base (contiguous 16)
  const int bklo = half * 16;   // B-operand per-lane K base (contiguous 16)

  f32x8 acc[4][4];
#pragma unroll
  for (int i = 0; i < 4; ++i)
#pragma unroll
    for (int t = 0; t < 4; ++t) acc[i][t] = fzero8();

  for (int kk = 0; kk < DD; kk += 32) {
    Frag16 a[4];
#pragma unroll
    for (int i = 0; i < 4; ++i) {
      const bf16* ap = A + (size_t)(m0 + i * 16 + l16) * DD + kk;
      a[i].h[0] = ld8(ap + aklo);
      a[i].h[1] = ld8(ap + 8 + aklo);
    }
#pragma unroll
    for (int t = 0; t < 4; ++t) {
      const bf16* bp = Bt + (size_t)(n0 + t * 16 + l16) * DD + kk + bklo;
      Frag16 b;
      b.h[0] = ld8(bp);
      b.h[1] = ld8(bp + 8);
#pragma unroll
      for (int i = 0; i < 4; ++i) acc[i][t] = wmma_bf16(a[i].v, b.v, acc[i][t]);
    }
  }

#pragma unroll
  for (int t = 0; t < 4; ++t) {
    const int n = n0 + t * 16 + l16;
    const int which = n >> 10;
    const int h = (n >> 6) & 15;
    const int j = n & 63;
#pragma unroll
    for (int i = 0; i < 4; ++i) {
#pragma unroll
      for (int r = 0; r < 8; ++r) {
        const int m = m0 + i * 16 + r + 8 * half;
        const int b = m >> 11;          // / SS
        const int s = m & (SS - 1);
        float val = acc[i][t][r];
        if (which == 0) {
          Qb[((size_t)((b * HH + h) * SS + s)) * HDIM + j] = (bf16)(val * 0.015625f);
        } else if (which == 1) {
          Kb[((size_t)((b * HH + h) * SS + s)) * HDIM + j] = (bf16)val;
        } else {
          Vt[((size_t)((b * HH + h) * HDIM + j)) * SS + s] = (bf16)val;
        }
      }
    }
  }
}

// ---------------------------------------------------------------- flash attention
// One wave per (b, h, 16-row query tile). Key blocks of 32, double-buffered
// into LDS with GLOBAL_LOAD_ASYNC_TO_LDS_B128 (ASYNCcnt), prefetching block
// t+1 while computing block t. 4x WMMA scores, online softmax, P relayout
// via LDS into A-operand form, 4x WMMA P*V.
__global__ __launch_bounds__(32) void attn_kernel(
    const bf16* __restrict__ Qb, const bf16* __restrict__ Kb,
    const bf16* __restrict__ Vt, bf16* __restrict__ Ac) {
#if USE_ASYNC
  __shared__ __align__(16) bf16 kstage[2][32 * HDIM];  // 2 x 4KB, flat copy of K rows
  __shared__ __align__(16) bf16 vstage[2][HDIM * 32];  // 2 x 4KB, rows of V^T
#endif
  __shared__ __align__(16) bf16 pbuf[16 * 32];
  const int lane = threadIdx.x & 31;
  const int half = lane >> 4;
  const int l16  = lane & 15;
  const int qt = blockIdx.x;
  const int mt = qt & (SS / 16 - 1);
  const int bh = qt >> 7;             // SS/16 == 128
  const int b = bh >> 4;
  const int h = bh & 15;

  const bf16* Qh = Qb + (size_t)bh * SS * HDIM;
  const bf16* Kh = Kb + (size_t)bh * SS * HDIM;
  const bf16* Vh = Vt + (size_t)bh * HDIM * SS;

  const int aklo = half * 16;
  const int bklo = half * 16;

  // Q fragments: k = 0..31 and 32..63
  const bf16* qp = Qh + (size_t)(mt * 16 + l16) * HDIM;
  Frag16 q0, q1;
  q0.h[0] = ld8(qp + aklo);       q0.h[1] = ld8(qp + 8 + aklo);
  q1.h[0] = ld8(qp + 32 + aklo);  q1.h[1] = ld8(qp + 40 + aklo);

  float mrun[8], lrun[8];
#pragma unroll
  for (int r = 0; r < 8; ++r) { mrun[r] = -1e30f; lrun[r] = 0.f; }
  f32x8 O[4] = {fzero8(), fzero8(), fzero8(), fzero8()};

#if USE_ASYNC
  // stage K tile (flat contiguous 4KB) + V tile (64 rows x 32 keys) for block t0
  auto stage_block = [&](int t0, int buf) {
    const bf16* kg = Kh + (size_t)t0 * HDIM;     // 2048 contiguous elements
#pragma unroll
    for (int j = 0; j < 8; ++j) {
      int e = (j * 32 + lane) * 8;               // element offset, 16B per lane
      async_cp_b128(kg + e, &kstage[buf][e]);
    }
#pragma unroll
    for (int j = 0; j < 8; ++j) {
      int c = j * 32 + lane;                     // 16B chunk id, 4 per V^T row
      int row = c >> 2;
      int cc  = c & 3;
      async_cp_b128(Vh + (size_t)row * SS + t0 + cc * 8, &vstage[buf][c * 8]);
    }
  };
  stage_block(0, 0);
#endif

  for (int t0 = 0; t0 < SS; t0 += 32) {
    Frag16 k00, k01, k10, k11;
#if USE_ASYNC
    const int buf = (t0 >> 5) & 1;
    if (t0 + 32 < SS) {
      // prior ds reads of buf^1 must be drained before overwriting it
      asm volatile("s_wait_dscnt 0" ::: "memory");
      stage_block(t0 + 32, buf ^ 1);
      asm volatile("s_wait_asynccnt 16" ::: "memory");  // current buf complete
    } else {
      asm volatile("s_wait_asynccnt 0" ::: "memory");
    }
    const bf16* kb = &kstage[buf][0];
    const bf16* vb = &vstage[buf][0];
    const bf16* kp0 = kb + (size_t)l16 * HDIM + bklo;
    const bf16* kp1 = kb + (size_t)(16 + l16) * HDIM + bklo;
#else
    const bf16* kp0 = Kh + (size_t)(t0 + l16) * HDIM + bklo;
    const bf16* kp1 = Kh + (size_t)(t0 + 16 + l16) * HDIM + bklo;
#endif
    k00.h[0] = ld8(kp0);       k00.h[1] = ld8(kp0 + 8);
    k01.h[0] = ld8(kp0 + 32);  k01.h[1] = ld8(kp0 + 40);
    k10.h[0] = ld8(kp1);       k10.h[1] = ld8(kp1 + 8);
    k11.h[0] = ld8(kp1 + 32);  k11.h[1] = ld8(kp1 + 40);

    f32x8 s0 = fzero8(), s1 = fzero8();
    s0 = wmma_bf16(q0.v, k00.v, s0);
    s0 = wmma_bf16(q1.v, k01.v, s0);
    s1 = wmma_bf16(q0.v, k10.v, s1);
    s1 = wmma_bf16(q1.v, k11.v, s1);

    float p0[8], p1[8], fac[8];
#pragma unroll
    for (int r = 0; r < 8; ++r) {
      float mc = fmaxf(s0[r], s1[r]);
      mc = fmaxf(mc, __shfl_xor(mc, 1, 32));
      mc = fmaxf(mc, __shfl_xor(mc, 2, 32));
      mc = fmaxf(mc, __shfl_xor(mc, 4, 32));
      mc = fmaxf(mc, __shfl_xor(mc, 8, 32));
      float mnew = fmaxf(mrun[r], mc);
      fac[r] = __expf(mrun[r] - mnew);
      p0[r] = __expf(s0[r] - mnew);
      p1[r] = __expf(s1[r] - mnew);
      float rs = p0[r] + p1[r];
      rs += __shfl_xor(rs, 1, 32);
      rs += __shfl_xor(rs, 2, 32);
      rs += __shfl_xor(rs, 4, 32);
      rs += __shfl_xor(rs, 8, 32);
      lrun[r] = lrun[r] * fac[r] + rs;
      mrun[r] = mnew;
    }
#pragma unroll
    for (int t = 0; t < 4; ++t)
#pragma unroll
      for (int r = 0; r < 8; ++r) O[t][r] *= fac[r];

    // Re-layout P (C layout) -> A-operand layout through LDS
    __syncthreads();
#pragma unroll
    for (int r = 0; r < 8; ++r) {
      int row = r + 8 * half;
      pbuf[row * 32 + l16]      = (bf16)p0[r];
      pbuf[row * 32 + 16 + l16] = (bf16)p1[r];
    }
    __syncthreads();
    Frag16 pa;
    pa.h[0] = ld8(&pbuf[l16 * 32 + aklo]);
    pa.h[1] = ld8(&pbuf[l16 * 32 + 8 + aklo]);

    // V fragments (B-operand rows = rows of V^T)
#pragma unroll
    for (int t = 0; t < 4; ++t) {
#if USE_ASYNC
      const bf16* vp = vb + (size_t)(t * 16 + l16) * 32 + bklo;
#else
      const bf16* vp = Vh + (size_t)(t * 16 + l16) * SS + t0 + bklo;
#endif
      Frag16 vfrag;
      vfrag.h[0] = ld8(vp);
      vfrag.h[1] = ld8(vp + 8);
      O[t] = wmma_bf16(pa.v, vfrag.v, O[t]);
    }
    __syncthreads();
  }

  // epilogue: attn / l, store head-concatenated [B,S,D] bf16
#pragma unroll
  for (int t = 0; t < 4; ++t) {
#pragma unroll
    for (int r = 0; r < 8; ++r) {
      int s = mt * 16 + r + 8 * half;
      float val = O[t][r] / lrun[r];
      Ac[((size_t)(b * SS + s)) * DD + h * HDIM + t * 16 + l16] = (bf16)val;
    }
  }
}

// ---------------------------------------------------------------- out projection
__global__ __launch_bounds__(256) void gemm_out_kernel(
    const bf16* __restrict__ A, const bf16* __restrict__ Bt, float* __restrict__ out) {
  const int lane = threadIdx.x & 31;
  const int wave = threadIdx.x >> 5;
  const int half = lane >> 4;
  const int l16  = lane & 15;
  const int m0 = blockIdx.x * 64;
  const int n0 = (blockIdx.y * 8 + wave) * 64;
  const int aklo = half * 16;
  const int bklo = half * 16;

  f32x8 acc[4][4];
#pragma unroll
  for (int i = 0; i < 4; ++i)
#pragma unroll
    for (int t = 0; t < 4; ++t) acc[i][t] = fzero8();

  for (int kk = 0; kk < DD; kk += 32) {
    Frag16 a[4];
#pragma unroll
    for (int i = 0; i < 4; ++i) {
      const bf16* ap = A + (size_t)(m0 + i * 16 + l16) * DD + kk;
      a[i].h[0] = ld8(ap + aklo);
      a[i].h[1] = ld8(ap + 8 + aklo);
    }
#pragma unroll
    for (int t = 0; t < 4; ++t) {
      const bf16* bp = Bt + (size_t)(n0 + t * 16 + l16) * DD + kk + bklo;
      Frag16 b;
      b.h[0] = ld8(bp);
      b.h[1] = ld8(bp + 8);
#pragma unroll
      for (int i = 0; i < 4; ++i) acc[i][t] = wmma_bf16(a[i].v, b.v, acc[i][t]);
    }
  }

#pragma unroll
  for (int t = 0; t < 4; ++t) {
    const int n = n0 + t * 16 + l16;
#pragma unroll
    for (int i = 0; i < 4; ++i) {
#pragma unroll
      for (int r = 0; r < 8; ++r) {
        const int m = m0 + i * 16 + r + 8 * half;
        out[(size_t)m * DD + n] = acc[i][t][r];
      }
    }
  }
}

// ---------------------------------------------------------------- launch
extern "C" void kernel_launch(void* const* d_in, const int* in_sizes, int n_in,
                              void* d_out, int out_size, void* d_ws, size_t ws_size,
                              hipStream_t stream) {
  const float* x  = (const float*)d_in[0];
  const float* Wq = (const float*)d_in[1];
  const float* Wk = (const float*)d_in[2];
  const float* Wv = (const float*)d_in[3];
  const float* Wo = (const float*)d_in[4];
  float* out = (float*)d_out;

  char* p = (char*)d_ws;
  bf16* xb    = (bf16*)p; p += (size_t)MM * DD * 2;       // 16 MB
  bf16* wqkvt = (bf16*)p; p += (size_t)3 * DD * DD * 2;   //  6 MB
  bf16* wot   = (bf16*)p; p += (size_t)DD * DD * 2;       //  2 MB
  bf16* Qb    = (bf16*)p; p += (size_t)MM * DD * 2;       // 16 MB
  bf16* Kb    = (bf16*)p; p += (size_t)MM * DD * 2;       // 16 MB
  bf16* Vt    = (bf16*)p; p += (size_t)MM * DD * 2;       // 16 MB
  bf16* Ac    = (bf16*)p; p += (size_t)MM * DD * 2;       // 16 MB

  cvt_x_kernel<<<(MM * DD + 255) / 256, 256, 0, stream>>>(x, xb, MM * DD);
  cvt_wqkv_kernel<<<(3 * HH * DD * HDIM + 255) / 256, 256, 0, stream>>>(Wq, Wk, Wv, wqkvt);
  cvt_wo_kernel<<<(DD * DD + 255) / 256, 256, 0, stream>>>(Wo, wot);

  gemm_qkv_kernel<<<dim3(MM / 64, (3 * DD) / 512), 256, 0, stream>>>(xb, wqkvt, Qb, Kb, Vt);
  attn_kernel<<<BB * HH * (SS / 16), 32, 0, stream>>>(Qb, Kb, Vt, Ac);
  gemm_out_kernel<<<dim3(MM / 64, DD / 512), 256, 0, stream>>>(Ac, wot, out);
}
